// SphericalVectorQuantizer_11020886081632
// MI455X (gfx1250) — compile-verified, matrix operand order checked by reading, not run on hardware
//
#include <hip/hip_runtime.h>
#include <hip/hip_bf16.h>
#include <stdint.h>

// ---------------------------------------------------------------------------
// Spherical VQ on MI455X (gfx1250, wave32, WMMA + async global->LDS)
//   scores = zf @ normalize(E)^T  -> argmax per row (streamed, never stored)
//   z_q    = normalize(E[argmax]) (fp32)
// Compute-bound (68.7 GFLOP vs ~24 MB of data) -> v_wmma_f32_16x16x32_bf16,
// with GLOBAL_LOAD_ASYNC_TO_LDS_B128 double-buffered staging (ASYNCcnt).
// ---------------------------------------------------------------------------

typedef __attribute__((ext_vector_type(16))) __bf16          v16bf;
typedef __attribute__((ext_vector_type(8)))  float           v8f;
typedef __attribute__((ext_vector_type(8)))  unsigned short  us8;
typedef __attribute__((ext_vector_type(16))) unsigned short  us16;

#define N_ROWS           8192      // 8 * 32 * 32
#define DIM              256
#define K_CODES          16384
#define KSPLIT           8
#define COLS_PER_SPLIT   (K_CODES / KSPLIT)          // 2048
#define CHUNK_N          64
#define CHUNKS_PER_SPLIT (COLS_PER_SPLIT / CHUNK_N)  // 32
#define M_TILE           128

// output layout in d_out (flat, reference return order, fp32)
#define OFF_IDX   2097152                      // after zq_out [8,256,32,32]
#define OFF_BITS  (OFF_IDX + N_ROWS)           // after indices [8192]
#define OFF_LOSS  (OFF_BITS + 2097152)         // after bit_indices [8,32,32,256]

__device__ __forceinline__ unsigned short f2bf(float f) {
  unsigned int u = __float_as_uint(f);
  unsigned int r = u + 0x7FFFu + ((u >> 16) & 1u);   // round-to-nearest-even
  return (unsigned short)(r >> 16);
}

__device__ __forceinline__ v16bf make_frag(const unsigned short* p) {
  // 16 bf16 = two 16B loads (K 0..7 / 16..23 of this lane's half)
  us8 lo = *(const us8*)(p);
  us8 hi = *(const us8*)(p + 16);
  us16 u;
#pragma unroll
  for (int j = 0; j < 8; ++j) { u[j] = lo[j]; u[j + 8] = hi[j]; }
  return __builtin_bit_cast(v16bf, u);
}

// gfx1250 async copy: 16B global -> LDS, tracked by ASYNCcnt (no VGPR bounce)
__device__ __forceinline__ void async_copy_b128(unsigned lds_addr,
                                                unsigned goff,
                                                unsigned long long gbase) {
  asm volatile("global_load_async_to_lds_b128 %0, %1, %2"
               :: "v"(lds_addr), "v"(goff), "s"(gbase) : "memory");
}
__device__ __forceinline__ void wait_async0() {
  asm volatile("s_wait_asynccnt 0" ::: "memory");
}

// ---- kernel 1: z [8,256,32,32] fp32 -> zf [8192,256] bf16 (row-major) -----
__global__ void pack_z_bf16(const float* __restrict__ z,
                            unsigned short* __restrict__ zfbf) {
  int t  = blockIdx.x * blockDim.x + threadIdx.x;  // 0 .. 2M-1
  int c  = t & 255;
  int n  = t >> 8;
  int b  = n >> 10;
  int hw = n & 1023;
  zfbf[t] = f2bf(z[((b * 256 + c) << 10) + hw]);
}

// ---- kernel 2: ew = normalize(E) -> bf16 [16384,256] ----------------------
__global__ void normalize_codes(const float* __restrict__ E,
                                unsigned short* __restrict__ ewbf) {
  __shared__ float red[256];
  const int r = blockIdx.x;
  const int c = threadIdx.x;
  float v = E[r * DIM + c];
  red[c] = v * v;
  __syncthreads();
  for (int s = 128; s > 0; s >>= 1) {
    if (c < s) red[c] += red[c + s];
    __syncthreads();
  }
  float denom = fmaxf(sqrtf(red[0]), 1e-12f);
  ewbf[r * DIM + c] = f2bf(v / denom);
}

// ---- kernel 3: streamed WMMA GEMM + running argmax ------------------------
// grid = (64 M-blocks, 8 K-splits), block = 256 threads = 8 waves.
// Each wave: 16 rows, A fragments (8 k-tiles) resident in VGPRs.
// Double-buffered 64-row codebook chunks staged with async global->LDS B128,
// overlapping HBM/L2 latency with the 4x8 WMMA stream per chunk.
__global__ void __launch_bounds__(256)
argmax_gemm(const unsigned short* __restrict__ zfbf,
            const unsigned short* __restrict__ ewbf,
            float* __restrict__ pvals, int* __restrict__ pidx) {
  __shared__ unsigned short Bs[2][CHUNK_N * DIM];  // 2 x 32 KB

  const int tid      = threadIdx.x;
  const int wave     = tid >> 5;
  const int lane     = tid & 31;
  const int halfsel  = lane >> 4;                  // 0: lanes 0-15, 1: 16-31
  const int l16      = lane & 15;
  const int mBase    = blockIdx.x * M_TILE + wave * 16;
  const int split    = blockIdx.y;
  const int colStart = split * COLS_PER_SPLIT;
  const int off0     = halfsel * 8;                // K-offset for this half

  const unsigned ldsBase[2] = {
      (unsigned)(uintptr_t)&Bs[0][0],              // flat addr[31:0] = LDS offset
      (unsigned)(uintptr_t)&Bs[1][0]
  };

  // A fragments: 16x32 bf16 per k-tile, row M = l16, K per ISA layout
  v16bf a[8];
  {
    const unsigned short* ap = zfbf + (mBase + l16) * DIM;
#pragma unroll
    for (int kk = 0; kk < 8; ++kk)
      a[kk] = make_frag(ap + kk * 32 + off0);
  }

  float bestV[8];
  int   bestI[8];
#pragma unroll
  for (int i = 0; i < 8; ++i) { bestV[i] = -3.0e38f; bestI[i] = 0x7fffffff; }

  // prologue: async-stage chunk 0 into buffer 0
  {
    const unsigned long long gbase =
        (unsigned long long)(uintptr_t)(ewbf + colStart * DIM);
#pragma unroll
    for (int i = 0; i < 8; ++i) {
      const unsigned e = (unsigned)(tid + i * 256) * 16u;  // byte offset
      async_copy_b128(ldsBase[0] + e, e, gbase);
    }
  }
  wait_async0();
  __syncthreads();

  for (int ch = 0; ch < CHUNKS_PER_SPLIT; ++ch) {
    const int cur       = ch & 1;
    const int chunkBase = colStart + ch * CHUNK_N;

    // async-stage next chunk into the other buffer (no wait yet)
    if (ch + 1 < CHUNKS_PER_SPLIT) {
      const unsigned long long gbase = (unsigned long long)(uintptr_t)(
          ewbf + (chunkBase + CHUNK_N) * DIM);
#pragma unroll
      for (int i = 0; i < 8; ++i) {
        const unsigned e = (unsigned)(tid + i * 256) * 16u;
        async_copy_b128(ldsBase[cur ^ 1] + e, e, gbase);
      }
    }

    // compute on the current buffer
    const unsigned short* bbuf = &Bs[cur][0];
#pragma unroll
    for (int t = 0; t < 4; ++t) {                // 4 n-tiles of 16 columns
      v8f acc = {0.f, 0.f, 0.f, 0.f, 0.f, 0.f, 0.f, 0.f};
#pragma unroll
      for (int kk = 0; kk < 8; ++kk) {           // 256 / 32
        v16bf b = make_frag(bbuf + (t * 16 + l16) * DIM + kk * 32 + off0);
        acc = __builtin_amdgcn_wmma_f32_16x16x32_bf16(
            /*neg_a=*/false, a[kk], /*neg_b=*/false, b,
            /*c_mod=*/(short)0, acc, /*reuse_a=*/false, /*reuse_b=*/false);
      }
      // D layout: VGPR i = row (i + 8*halfsel), lane l16 = column
      const int col = chunkBase + t * 16 + l16;
#pragma unroll
      for (int i = 0; i < 8; ++i)
        if (acc[i] > bestV[i]) { bestV[i] = acc[i]; bestI[i] = col; }
    }

    // own async loads done, then all waves' loads + reads of cur buffer done
    wait_async0();
    __syncthreads();
  }

  // cross-lane argmax within each 16-lane half (ties -> smallest index,
  // matching jnp.argmin's first-occurrence rule)
#pragma unroll
  for (int i = 0; i < 8; ++i) {
    float v  = bestV[i];
    int   ix = bestI[i];
#pragma unroll
    for (int m = 8; m >= 1; m >>= 1) {
      float ov = __shfl_xor(v, m, 32);
      int   oi = __shfl_xor(ix, m, 32);
      if (ov > v || (ov == v && oi < ix)) { v = ov; ix = oi; }
    }
    bestV[i] = v; bestI[i] = ix;
  }
  if (l16 == 0) {
#pragma unroll
    for (int i = 0; i < 8; ++i) {
      const int row = mBase + halfsel * 8 + i;
      pvals[row * KSPLIT + split] = bestV[i];
      pidx [row * KSPLIT + split] = bestI[i];
    }
  }
}

// ---- kernel 4: combine K-splits ------------------------------------------
__global__ void reduce_splits(const float* __restrict__ pvals,
                              const int* __restrict__ pidx,
                              int* __restrict__ indices) {
  int n = blockIdx.x * blockDim.x + threadIdx.x;
  if (n >= N_ROWS) return;
  float bv = pvals[n * KSPLIT];
  int   bi = pidx [n * KSPLIT];
  for (int s = 1; s < KSPLIT; ++s) {
    float v  = pvals[n * KSPLIT + s];
    int   ix = pidx [n * KSPLIT + s];
    if (v > bv || (v == bv && ix < bi)) { bv = v; bi = ix; }
  }
  indices[n] = bi;
}

// ---- kernel 5: gather + fp32 normalize + all outputs ----------------------
__global__ void gather_outputs(const float* __restrict__ E,
                               const int* __restrict__ indices,
                               float* __restrict__ out) {
  __shared__ float red[256];
  const int n   = blockIdx.x;
  const int c   = threadIdx.x;
  const int idx = indices[n];
  float v = E[idx * DIM + c];
  red[c] = v * v;
  __syncthreads();
  for (int s = 128; s > 0; s >>= 1) {
    if (c < s) red[c] += red[c + s];
    __syncthreads();
  }
  float denom = fmaxf(sqrtf(red[0]), 1e-12f);
  float q = v / denom;

  const int b = n >> 10, hw = n & 1023;
  out[((b * 256 + c) << 10) + hw] = q;                               // zq [b,d,h,w]
  out[OFF_BITS + n * DIM + c] =
      (float)((int)(q * 5.656854249492381f) + 4);                    // trunc-to-zero
  if (c == 0) out[OFF_IDX + n] = (float)idx;
  if (n == 0 && c == 0) out[OFF_LOSS] = 0.0f;
}

// ---------------------------------------------------------------------------
extern "C" void kernel_launch(void* const* d_in, const int* in_sizes, int n_in,
                              void* d_out, int out_size, void* d_ws, size_t ws_size,
                              hipStream_t stream) {
  (void)in_sizes; (void)n_in; (void)out_size; (void)ws_size;
  const float* z = (const float*)d_in[0];        // [8,256,32,32]
  const float* E = (const float*)d_in[1];        // [16384,256]
  float* out = (float*)d_out;

  char* ws = (char*)d_ws;
  unsigned short* zfbf = (unsigned short*)(ws);                       // 4 MiB
  unsigned short* ewbf = (unsigned short*)(ws + (4u  << 20));         // 8 MiB
  float* pvals   = (float*)(ws + (12u << 20));                        // 256 KiB
  int*   pidx    = (int*)  (ws + (12u << 20) + N_ROWS * KSPLIT * 4);  // 256 KiB
  int*   indices = (int*)  (ws + (12u << 20) + 2 * N_ROWS * KSPLIT * 4);

  pack_z_bf16    <<<N_ROWS, 256, 0, stream>>>(z, zfbf);
  normalize_codes<<<K_CODES, 256, 0, stream>>>(E, ewbf);
  dim3 g(N_ROWS / M_TILE, KSPLIT);               // 64 x 8 blocks
  argmax_gemm    <<<g, 256, 0, stream>>>(zfbf, ewbf, pvals, pidx);
  reduce_splits  <<<N_ROWS / 256, 256, 0, stream>>>(pvals, pidx, indices);
  gather_outputs <<<N_ROWS, 256, 0, stream>>>(E, indices, out);
}
